// AstAttendGruV3Decoder_80023830659286
// MI455X (gfx1250) — matrix operand order repaired and checked by reading, not compiled, not run on hardware
//
#include <hip/hip_runtime.h>
#include <hip/hip_bf16.h>

// ---------------------------------------------------------------------------
// AstAttendGruV3 decoder for MI455X (gfx1250, wave32).
//   k_gh      : gh = h0 @ W_hh.T + b_hh                    (64 x 768)
//   k_dec     : gi = wemb[tok] @ W_ih.T + b_ih ; GRU gates -> dec (1920 x 256)
//   k_attn    : two attentions (reshape-keys quirk!) + cat @ W1.T  (1920 x 256)
//   k_gemm    : relu(out1 @ W2.T + b2) -> raw logits, WMMA bf16x3  (1920 x 50000)
//   k_softmax : per-row online logsumexp + argmax, in-place logprobs
// Memory-bound (~1.5 GB traffic @ 23.3 TB/s); WMMA compute is ~free.
// ---------------------------------------------------------------------------

#define B_   64
#define T_   30
#define H_   256
#define S_   100
#define V_   50000
#define G3H  768
#define AP   257   // padded LDS row stride (bank-conflict avoidance, 64 banks)

typedef __attribute__((ext_vector_type(16))) __bf16 v16bf;
typedef __attribute__((ext_vector_type(8)))  float  v8f;

// ---- gfx1250 async global->LDS staging (guarded; falls back to plain copy) --
#if defined(__has_builtin)
#if __has_builtin(__builtin_amdgcn_global_load_async_to_lds_b32)
#define HAVE_ASYNC_LDS 1
#endif
#endif
#ifndef HAVE_ASYNC_LDS
#define HAVE_ASYNC_LDS 0
#endif

#if HAVE_ASYNC_LDS
typedef __attribute__((address_space(1))) int glob_i32;   // "__device__ int"
typedef __attribute__((address_space(3))) int lds_i32;    // "__shared__ int"
#endif

__device__ __forceinline__ void lds_copy_f32(const float* g, float* l) {
#if HAVE_ASYNC_LDS
  __builtin_amdgcn_global_load_async_to_lds_b32(
      (glob_i32*)g, (lds_i32*)l, 0, 0);
#else
  *l = *g;
#endif
}

__device__ __forceinline__ void lds_copy_wait() {
#if HAVE_ASYNC_LDS
#if __has_builtin(__builtin_amdgcn_s_wait_asynccnt)
  __builtin_amdgcn_s_wait_asynccnt(0);
#else
  asm volatile("s_wait_asynccnt 0x0" ::: "memory");
#endif
#endif
}

// ---------------------------------------------------------------------------
// gh[b][o] = b_hh[o] + dot(h0[b], W_hh[o])        grid=64, block=256
// ---------------------------------------------------------------------------
__global__ void __launch_bounds__(256)
k_gh(const float* __restrict__ h0, const float* __restrict__ Whh,
     const float* __restrict__ bhh, float* __restrict__ gh) {
  __shared__ float sh[H_];
  const int b = blockIdx.x, tid = threadIdx.x;
  lds_copy_f32(h0 + (size_t)b * H_ + tid, &sh[tid]);
  lds_copy_wait();
  __syncthreads();
#pragma unroll
  for (int g = 0; g < 3; ++g) {
    const int o = g * H_ + tid;
    const float* wr = Whh + (size_t)o * H_;
    float acc = bhh[o];
    for (int k = 0; k < H_; ++k) acc += sh[k] * wr[k];
    gh[(size_t)b * G3H + o] = acc;
  }
}

// ---------------------------------------------------------------------------
// GRU gate math per (b,t) row.                     grid=1920, block=256
// ---------------------------------------------------------------------------
__global__ void __launch_bounds__(256)
k_dec(const int* __restrict__ cin, const float* __restrict__ wemb,
      const float* __restrict__ Wih, const float* __restrict__ bih,
      const float* __restrict__ gh, const float* __restrict__ h0,
      float* __restrict__ dec) {
  __shared__ float se[H_];
  __shared__ float sgh[G3H];
  const int row = blockIdx.x, tid = threadIdx.x;
  const int b = row / T_;
  const int tok = cin[row];                      // comment_input[b][t]
  lds_copy_f32(wemb + (size_t)tok * H_ + tid, &se[tid]);
#pragma unroll
  for (int g = 0; g < 3; ++g)
    lds_copy_f32(gh + (size_t)b * G3H + g * H_ + tid, &sgh[g * H_ + tid]);
  lds_copy_wait();
  __syncthreads();

  const int h = tid;
  float gir = bih[h], giz = bih[H_ + h], gin = bih[2 * H_ + h];
  const float* wr = Wih + (size_t)h * H_;
  const float* wz = Wih + (size_t)(H_ + h) * H_;
  const float* wn = Wih + (size_t)(2 * H_ + h) * H_;
  for (int k = 0; k < H_; ++k) {
    const float e = se[k];
    gir += e * wr[k]; giz += e * wz[k]; gin += e * wn[k];
  }
  const float r = 1.f / (1.f + expf(-(gir + sgh[h])));
  const float z = 1.f / (1.f + expf(-(giz + sgh[H_ + h])));
  const float n = tanhf(gin + r * sgh[2 * H_ + h]);
  dec[(size_t)row * H_ + h] = (1.f - z) * n + z * h0[(size_t)b * H_ + h];
}

// ---------------------------------------------------------------------------
// Attention (keys = enc reshaped (H,S), NOT transposed!) + cat @ W1.T + relu.
// grid=1920, block=256. Threads 0..99 score tok, 128..227 score ast.
// ---------------------------------------------------------------------------
__global__ void __launch_bounds__(256)
k_attn(const float* __restrict__ tok, const float* __restrict__ ast,
       const float* __restrict__ decg, const float* __restrict__ W1,
       const float* __restrict__ b1, float* __restrict__ out1) {
  __shared__ float sd[H_];
  __shared__ float sa[2][S_];
  __shared__ float cat[G3H];
  __shared__ float red[2][2];
  const int row = blockIdx.x, tid = threadIdx.x;
  const int b = row / T_;
  lds_copy_f32(decg + (size_t)row * H_ + tid, &sd[tid]);
  lds_copy_wait();
  __syncthreads();

  const float* e0 = tok + (size_t)b * S_ * H_;
  const float* e1 = ast + (size_t)b * S_ * H_;
  if (tid < S_) {                                  // score[s] = sum_h sd[h]*flat[h*S+s]
    float acc = 0.f;
    for (int h = 0; h < H_; ++h) acc += sd[h] * e0[h * S_ + tid];
    sa[0][tid] = acc;
  } else if (tid >= 128 && tid < 128 + S_) {
    const int s = tid - 128;
    float acc = 0.f;
    for (int h = 0; h < H_; ++h) acc += sd[h] * e1[h * S_ + s];
    sa[1][s] = acc;
  }
  __syncthreads();
  if (tid == 0 || tid == 128) {                    // softmax stats over S=100
    const int e = tid >> 7;
    float m = sa[e][0];
    for (int s = 1; s < S_; ++s) m = fmaxf(m, sa[e][s]);
    float sum = 0.f;
    for (int s = 0; s < S_; ++s) sum += expf(sa[e][s] - m);
    red[e][0] = m; red[e][1] = 1.f / sum;
  }
  __syncthreads();
  if (tid < S_)                       sa[0][tid]     = expf(sa[0][tid]     - red[0][0]) * red[0][1];
  else if (tid >= 128 && tid < 228) { const int s = tid - 128;
                                      sa[1][s]       = expf(sa[1][s]       - red[1][0]) * red[1][1]; }
  __syncthreads();
  {                                                // ctx uses enc row-major normally
    const int h = tid;
    float c0 = 0.f, c1 = 0.f;
    for (int s = 0; s < S_; ++s) {
      c0 += sa[0][s] * e0[s * H_ + h];
      c1 += sa[1][s] * e1[s * H_ + h];
    }
    cat[h] = c0; cat[H_ + h] = sd[h]; cat[2 * H_ + h] = c1;
  }
  __syncthreads();
  {
    const int h = tid;
    const float* wr = W1 + (size_t)h * G3H;
    float acc = b1[h];
    for (int j = 0; j < G3H; ++j) acc += cat[j] * wr[j];
    out1[(size_t)row * H_ + h] = fmaxf(acc, 0.f);
  }
}

// ---------------------------------------------------------------------------
// Big GEMM: logits = relu(out1 @ W2.T + b2), bf16x3 WMMA (near-fp32 accuracy).
// block = 160 thr = 5 waves; per block: M=16 rows, N=80 cols (16/wave), K=256.
// grid = (50000/80, 1920/16) = (625, 120).
// A staged in LDS (async); B fragments are contiguous 64B runs of W2 rows
// (lane layout: component c <-> K = koff + c), loaded straight from global
// (W2 = 51 MB, resident in the 192 MB L2 across all 120 M-tiles).
// ---------------------------------------------------------------------------
__global__ void __launch_bounds__(160)
k_gemm(const float* __restrict__ Aout1, const float* __restrict__ W2,
       const float* __restrict__ b2, float* __restrict__ logits) {
  __shared__ float At[16 * AP];
  const int tid  = threadIdx.x;
  const int wave = tid >> 5, lane = tid & 31;
  const int m0 = blockIdx.y * 16, n0 = blockIdx.x * 80;

  for (int idx = tid; idx < 16 * H_; idx += 160) {
    const int i = idx >> 8, k = idx & 255;
    lds_copy_f32(Aout1 + (size_t)(m0 + i) * H_ + k, &At[i * AP + k]);
  }
  lds_copy_wait();
  __syncthreads();

  const int mm = lane & 15, hw = lane >> 4;        // A: M=mm, K-half select
  const int ncol = n0 + wave * 16 + (lane & 15);   // B/D: N column
  const float* w2row = W2 + (size_t)ncol * H_;
  v8f acc = {};

#pragma unroll
  for (int ks = 0; ks < 8; ++ks) {
    const int kb = ks * 32;
    v16bf ahi, alo, bhi, blo;
    // A 16x32 bf16 fragment: comp c -> K = kb+hw*16+c (contiguous)
    const int ka = kb + hw * 16;
#pragma unroll
    for (int i = 0; i < 8; ++i) {
      float f = At[mm * AP + ka + i];
      __bf16 fh = (__bf16)f;
      ahi[i] = fh; alo[i] = (__bf16)(f - (float)fh);
      float f2 = At[mm * AP + ka + 8 + i];
      __bf16 fh2 = (__bf16)f2;
      ahi[8 + i] = fh2; alo[8 + i] = (__bf16)(f2 - (float)fh2);
    }
    // B 32x16 bf16 fragment: lane n=ncol, K = kb + hw*16 + {0..15}, contiguous
    const float4* wq = (const float4*)(w2row + kb + hw * 16);
    const float4 q0 = wq[0], q1 = wq[1], q2 = wq[2], q3 = wq[3];
    float fl[16];
    fl[0]=q0.x;  fl[1]=q0.y;  fl[2]=q0.z;  fl[3]=q0.w;
    fl[4]=q1.x;  fl[5]=q1.y;  fl[6]=q1.z;  fl[7]=q1.w;
    fl[8]=q2.x;  fl[9]=q2.y;  fl[10]=q2.z; fl[11]=q2.w;
    fl[12]=q3.x; fl[13]=q3.y; fl[14]=q3.z; fl[15]=q3.w;
#pragma unroll
    for (int i = 0; i < 16; ++i) {
      const float f = fl[i];
      const __bf16 fh = (__bf16)f;
      bhi[i] = fh; blo[i] = (__bf16)(f - (float)fh);
    }
    // bf16x3: AhiBhi + AloBhi + AhiBlo, f32 accumulate
    acc = __builtin_amdgcn_wmma_f32_16x16x32_bf16(false, ahi, false, bhi, (short)0, acc, false, false);
    acc = __builtin_amdgcn_wmma_f32_16x16x32_bf16(false, alo, false, bhi, (short)0, acc, false, false);
    acc = __builtin_amdgcn_wmma_f32_16x16x32_bf16(false, ahi, false, blo, (short)0, acc, false, false);
  }

  const float bias = b2[ncol];
#pragma unroll
  for (int j = 0; j < 8; ++j) {                    // D: VGPR j -> M = j + hw*8
    const int r = m0 + j + hw * 8;
    logits[(size_t)r * V_ + ncol] = fmaxf(acc[j] + bias, 0.f);
  }
}

// ---------------------------------------------------------------------------
// Per-row fused online logsumexp + first-occurrence argmax, in-place subtract.
// grid=1920, block=256. Second pass is L2-warm (row = 200 KB).
// ---------------------------------------------------------------------------
__global__ void __launch_bounds__(256)
k_softmax(float* __restrict__ logits, float* __restrict__ seq) {
  __shared__ float rm[256], rs[256];
  __shared__ int   ri[256];
  const int row = blockIdx.x, tid = threadIdx.x;
  float* p = logits + (size_t)row * V_;

  float m = -1e30f, s = 0.f; int idx = 0;
  for (int i = tid; i < V_; i += 256) {
    __builtin_prefetch(p + i + 4096, 0, 1);        // global_prefetch_b8
    const float v = p[i];
    if (v > m) { s = s * expf(m - v) + 1.f; m = v; idx = i; }
    else       { s += expf(v - m); }
  }
  rm[tid] = m; rs[tid] = s; ri[tid] = idx;
  __syncthreads();
  for (int off = 128; off > 0; off >>= 1) {
    if (tid < off) {
      const float m1 = rm[tid], s1 = rs[tid];       const int i1 = ri[tid];
      const float m2 = rm[tid + off], s2 = rs[tid + off]; const int i2 = ri[tid + off];
      if (m2 > m1)       { rm[tid] = m2; rs[tid] = s1 * expf(m1 - m2) + s2; ri[tid] = i2; }
      else if (m2 == m1) { rs[tid] = s1 + s2; ri[tid] = (i1 < i2) ? i1 : i2; }
      else               { rs[tid] = s1 + s2 * expf(m2 - m1); }
    }
    __syncthreads();
  }
  const float logZ = rm[0] + logf(rs[0]);
  if (tid == 0) seq[row] = (float)ri[0];
  for (int i = tid; i < V_; i += 256) p[i] -= logZ;
}

// ---------------------------------------------------------------------------
extern "C" void kernel_launch(void* const* d_in, const int* in_sizes, int n_in,
                              void* d_out, int out_size, void* d_ws, size_t ws_size,
                              hipStream_t stream) {
  (void)in_sizes; (void)n_in; (void)out_size; (void)ws_size;
  const int*   cin  = (const int*)  d_in[1];   // comment_input (64,30)
  const float* tok  = (const float*)d_in[2];   // tok_output   (64,100,256)
  const float* h0   = (const float*)d_in[3];   // tok_enc_hc[0] (64,256)
  const float* ast  = (const float*)d_in[4];   // sbtao_output (64,100,256)
  const float* wemb = (const float*)d_in[5];
  const float* Wih  = (const float*)d_in[6];
  const float* Whh  = (const float*)d_in[7];
  const float* bih  = (const float*)d_in[8];
  const float* bhh  = (const float*)d_in[9];
  const float* W1   = (const float*)d_in[10];
  const float* b1   = (const float*)d_in[11];
  const float* W2   = (const float*)d_in[12];
  const float* b2   = (const float*)d_in[13];

  float* ws   = (float*)d_ws;
  float* gh   = ws;                              // 64*768
  float* dec  = ws + (size_t)B_ * G3H;           // 1920*256
  float* out1 = dec + (size_t)B_ * T_ * H_;      // 1920*256

  float* o      = (float*)d_out;
  float* seq    = o;                             // 1920 (argmax as float)
  float* logits = o + (size_t)B_ * T_;           // 1920*50000

  k_gh     <<<B_,            256, 0, stream>>>(h0, Whh, bhh, gh);
  k_dec    <<<B_ * T_,       256, 0, stream>>>(cin, wemb, Wih, bih, gh, h0, dec);
  k_attn   <<<B_ * T_,       256, 0, stream>>>(tok, ast, dec, W1, b1, out1);
  k_gemm   <<<dim3(V_ / 80, (B_ * T_) / 16), 160, 0, stream>>>(out1, W2, b2, logits);
  k_softmax<<<B_ * T_,       256, 0, stream>>>(logits, seq);
}
